// CharVanillaRNN_18184891531992
// MI455X (gfx1250) — matrix operand order repaired and checked by reading, not compiled, not run on hardware
//
#include <hip/hip_runtime.h>
#include <math.h>

// ---------------------------------------------------------------------------
// CharVanillaRNN on MI455X (gfx1250, wave32, WMMA bf16 16x16x32, f32 accum)
//
// VOCAB=128, EMB=256, HID=1024, B=256, S=256
//
// Strategy:
//  * Precompute P0[VOCAB,HID] = emb @ W_ih0^T + b_ih0  (f32, 512KB) -> layer-0
//    input projection becomes a gather.
//  * Per step t (stream-ordered, graph-captured):
//      L0: h1' = tanh(P0[x[:,t]] + h1 @ W_hh0^T)
//      L1: h2' = tanh(h1' @ W_ih1^T + b_ih1 + h2 @ W_hh1^T)   (also -> outs[t])
//    Each GEMM: [256,1024] x [1024,1024], bf16 in / f32 accumulate via
//    v_wmma_f32_16x16x32_bf16. Weights (6MB bf16) are L2-resident.
//  * Logits: [S*B=65536,1024] @ fc_w^T[1024,128] + fc_b, scatter to [B,S,V].
// ---------------------------------------------------------------------------

#define VOCAB 128
#define EMB   256
#define HID   1024
#define BATCH 256
#define SEQ   256

typedef __bf16 bf16_t;
typedef __attribute__((ext_vector_type(16))) __bf16 v16bf;
typedef __attribute__((ext_vector_type(8)))  __bf16 v8bf;
typedef __attribute__((ext_vector_type(8)))  float  v8f;

union V16 { v16bf v; v8bf h[2]; };

// ------------------------------ prep kernels -------------------------------

__global__ __launch_bounds__(256) void cvt_f32_bf16_kernel(
    const float* __restrict__ src, bf16_t* __restrict__ dst, int n) {
  int i = blockIdx.x * blockDim.x + threadIdx.x;
  int stride = gridDim.x * blockDim.x;
  for (; i < n; i += stride) dst[i] = (bf16_t)src[i];
}

__global__ __launch_bounds__(256) void fill_zero_bf16_kernel(
    bf16_t* __restrict__ dst, int n) {
  int i = blockIdx.x * blockDim.x + threadIdx.x;
  int stride = gridDim.x * blockDim.x;
  for (; i < n; i += stride) dst[i] = (bf16_t)0.0f;
}

__global__ __launch_bounds__(256) void cvt_bf16_f32_kernel(
    const bf16_t* __restrict__ src, float* __restrict__ dst, int n) {
  int i = blockIdx.x * blockDim.x + threadIdx.x;
  int stride = gridDim.x * blockDim.x;
  for (; i < n; i += stride) dst[i] = (float)src[i];
}

// P0[v][h] = b_ih0[h] + sum_e emb[v][e] * W_ih0[h][e]     (128*1024 outputs)
__global__ __launch_bounds__(256) void p0_kernel(
    const float* __restrict__ emb, const float* __restrict__ Wih0,
    const float* __restrict__ b0, float* __restrict__ P0) {
  int t = blockIdx.x * blockDim.x + threadIdx.x;   // 0..131071
  if (t >= VOCAB * HID) return;
  int v = t >> 10;          // / HID
  int h = t & (HID - 1);
  const float* e = emb + (size_t)v * EMB;
  const float* w = Wih0 + (size_t)h * EMB;
  float s = b0[h];
  for (int k = 0; k < EMB; ++k) s = fmaf(e[k], w[k], s);
  P0[t] = s;
}

// ------------------------------ RNN step GEMM ------------------------------
// OUT[256,1024] = tanh( addend + A1 @ W1^T (+ A2 @ W2^T) )
//   A1/A2: [256,1024] bf16 (row-major), W1/W2: [1024,1024] bf16 (row-major)
//   addend: P0 gather (layer 0) or bias (layer 1)
// Block: 256 threads = 8 waves, 2(M) x 4(N); wave tile = 16M x 64N.
// Grid: (4, 8) -> block tile 32M x 256N covers 256 x 1024.

__global__ __launch_bounds__(256) void rnn_step_kernel(
    const bf16_t* __restrict__ A1, const bf16_t* __restrict__ W1,
    const bf16_t* __restrict__ A2, const bf16_t* __restrict__ W2,
    const float*  __restrict__ P0,    // [VOCAB,HID] f32, or null
    const int*    __restrict__ tok,   // &x[t]; token of row b = tok[b*SEQ]
    const float*  __restrict__ bias,  // [HID] f32, or null
    bf16_t* __restrict__ Hout,        // [256,1024] bf16
    bf16_t* __restrict__ Oout)        // optional duplicate store (outs slot)
{
  const int lane  = threadIdx.x & 31;
  const int wave  = threadIdx.x >> 5;
  const int waveM = wave >> 2;        // 0..1
  const int waveN = wave & 3;         // 0..3
  const int M0 = blockIdx.y * 32 + waveM * 16;
  const int N0 = blockIdx.x * 256 + waveN * 64;

  const int lmod = lane & 15;
  const int lhi  = lane >> 4;         // 0 | 1
  const int akoff = lhi * 8;          // A frag: K halves {k0..k0+7, k0+16..k0+23}
  const int bkoff = lhi * 8;          // B frag: K halves {k0..k0+7, k0+16..k0+23}

  v8f zero = {0.f, 0.f, 0.f, 0.f, 0.f, 0.f, 0.f, 0.f};
  v8f acc[4] = {zero, zero, zero, zero};

  const int arow = M0 + lmod;
  const bf16_t* a1p = A1 + (size_t)arow * HID;

  // pass 1: A1 @ W1^T
  for (int kb = 0; kb < HID; kb += 32) {
    V16 a;
    a.h[0] = *(const v8bf*)(a1p + kb + akoff);
    a.h[1] = *(const v8bf*)(a1p + kb + akoff + 16);
#pragma unroll
    for (int j = 0; j < 4; ++j) {
      const int n = N0 + j * 16 + lmod;
      const bf16_t* bp = W1 + (size_t)n * HID + kb + bkoff;
      V16 b;
      b.h[0] = *(const v8bf*)(bp);
      b.h[1] = *(const v8bf*)(bp + 16);
      acc[j] = __builtin_amdgcn_wmma_f32_16x16x32_bf16(
          false, a.v, false, b.v, (short)0, acc[j], false, false);
    }
  }

  // pass 2 (layer 1 only): A2 @ W2^T
  if (A2) {
    const bf16_t* a2p = A2 + (size_t)arow * HID;
    for (int kb = 0; kb < HID; kb += 32) {
      V16 a;
      a.h[0] = *(const v8bf*)(a2p + kb + akoff);
      a.h[1] = *(const v8bf*)(a2p + kb + akoff + 16);
#pragma unroll
      for (int j = 0; j < 4; ++j) {
        const int n = N0 + j * 16 + lmod;
        const bf16_t* bp = W2 + (size_t)n * HID + kb + bkoff;
        V16 b;
        b.h[0] = *(const v8bf*)(bp);
        b.h[1] = *(const v8bf*)(bp + 16);
        acc[j] = __builtin_amdgcn_wmma_f32_16x16x32_bf16(
            false, a.v, false, b.v, (short)0, acc[j], false, false);
      }
    }
  }

  // epilogue: C/D layout -> lane: N = lane%16; VGPR r: M = r + 8*(lane>=16)
#pragma unroll
  for (int j = 0; j < 4; ++j) {
    const int col = N0 + j * 16 + lmod;
    const float badd = bias ? bias[col] : 0.0f;
#pragma unroll
    for (int r = 0; r < 8; ++r) {
      const int row = M0 + r + lhi * 8;
      float v = acc[j][r] + badd;
      if (P0) {
        const int tk = tok[row * SEQ];
        v += P0[(size_t)tk * HID + col];
      }
      v = tanhf(v);
      const bf16_t bv = (bf16_t)v;
      Hout[(size_t)row * HID + col] = bv;
      if (Oout) Oout[(size_t)row * HID + col] = bv;
    }
  }
}

// ------------------------------ logits GEMM --------------------------------
// logits[b][s][v] = outs_row(m = s*B + b) . fc_w[v] + fc_b[v]
// Block: 8 waves, 4(M) x 2(N); wave tile 16x64; block tile 64M x 128N.
// Grid: (1, 1024) -> M = 65536, N = 128, K = 1024.

__global__ __launch_bounds__(256) void logits_kernel(
    const bf16_t* __restrict__ A,    // outs [S*B, HID] bf16
    const bf16_t* __restrict__ W,    // fc_w [VOCAB, HID] bf16
    const float*  __restrict__ bias, // fc_b [VOCAB] f32
    float* __restrict__ out)         // logits [B, S, VOCAB] f32
{
  const int lane  = threadIdx.x & 31;
  const int wave  = threadIdx.x >> 5;
  const int waveM = wave >> 1;        // 0..3
  const int waveN = wave & 1;         // 0..1
  const int M0 = blockIdx.y * 64 + waveM * 16;
  const int N0 = waveN * 64;

  const int lmod = lane & 15;
  const int lhi  = lane >> 4;
  const int akoff = lhi * 8;
  const int bkoff = lhi * 8;

  v8f zero = {0.f, 0.f, 0.f, 0.f, 0.f, 0.f, 0.f, 0.f};
  v8f acc[4] = {zero, zero, zero, zero};

  const int arow = M0 + lmod;
  const bf16_t* ap = A + (size_t)arow * HID;

  for (int kb = 0; kb < HID; kb += 32) {
    V16 a;
    a.h[0] = *(const v8bf*)(ap + kb + akoff);
    a.h[1] = *(const v8bf*)(ap + kb + akoff + 16);
#pragma unroll
    for (int j = 0; j < 4; ++j) {
      const int n = N0 + j * 16 + lmod;
      const bf16_t* bp = W + (size_t)n * HID + kb + bkoff;
      V16 b;
      b.h[0] = *(const v8bf*)(bp);
      b.h[1] = *(const v8bf*)(bp + 16);
      acc[j] = __builtin_amdgcn_wmma_f32_16x16x32_bf16(
          false, a.v, false, b.v, (short)0, acc[j], false, false);
    }
  }

#pragma unroll
  for (int j = 0; j < 4; ++j) {
    const int n = N0 + j * 16 + lmod;
    const float badd = bias[n];
#pragma unroll
    for (int r = 0; r < 8; ++r) {
      const int m = M0 + r + lhi * 8;   // m = s*B + b
      const int s = m >> 8;             // / BATCH
      const int b = m & 255;
      out[((size_t)b * SEQ + s) * VOCAB + n] = acc[j][r] + badd;
    }
  }
}

// ------------------------------- launcher ----------------------------------

extern "C" void kernel_launch(void* const* d_in, const int* in_sizes, int n_in,
                              void* d_out, int out_size, void* d_ws, size_t ws_size,
                              hipStream_t stream) {
  (void)in_sizes; (void)n_in; (void)out_size; (void)ws_size;

  const int*   x    = (const int*)d_in[0];    // [B,S]
  const float* emb  = (const float*)d_in[1];  // [VOCAB,EMB]
  const float* Wih0 = (const float*)d_in[2];  // [HID,EMB]
  const float* bih0 = (const float*)d_in[3];  // [HID]
  const float* Whh0 = (const float*)d_in[4];  // [HID,HID]
  const float* Wih1 = (const float*)d_in[5];  // [HID,HID]
  const float* bih1 = (const float*)d_in[6];  // [HID]
  const float* Whh1 = (const float*)d_in[7];  // [HID,HID]
  const float* fcw  = (const float*)d_in[8];  // [VOCAB,HID]
  const float* fcb  = (const float*)d_in[9];  // [VOCAB]
  float* out = (float*)d_out;                 // logits + h1_f + h2_f

  // ---- workspace carve (~138 MB) ----
  char* ws = (char*)d_ws;
  auto carve = [&](size_t bytes) -> char* {
    char* p = ws;
    ws += (bytes + 255) & ~(size_t)255;
    return p;
  };
  bf16_t* Whh0b = (bf16_t*)carve((size_t)HID * HID * 2);
  bf16_t* Wih1b = (bf16_t*)carve((size_t)HID * HID * 2);
  bf16_t* Whh1b = (bf16_t*)carve((size_t)HID * HID * 2);
  bf16_t* fcwb  = (bf16_t*)carve((size_t)VOCAB * HID * 2);
  float*  P0f   = (float*) carve((size_t)VOCAB * HID * 4);
  bf16_t* h1a   = (bf16_t*)carve((size_t)BATCH * HID * 2);
  bf16_t* h1b   = (bf16_t*)carve((size_t)BATCH * HID * 2);
  bf16_t* h2a   = (bf16_t*)carve((size_t)BATCH * HID * 2);
  bf16_t* h2b   = (bf16_t*)carve((size_t)BATCH * HID * 2);
  bf16_t* outs  = (bf16_t*)carve((size_t)SEQ * BATCH * HID * 2);  // 128 MB

  // ---- prep ----
  cvt_f32_bf16_kernel<<<1024, 256, 0, stream>>>(Whh0, Whh0b, HID * HID);
  cvt_f32_bf16_kernel<<<1024, 256, 0, stream>>>(Wih1, Wih1b, HID * HID);
  cvt_f32_bf16_kernel<<<1024, 256, 0, stream>>>(Whh1, Whh1b, HID * HID);
  cvt_f32_bf16_kernel<<<512,  256, 0, stream>>>(fcw, fcwb, VOCAB * HID);
  p0_kernel<<<(VOCAB * HID) / 256, 256, 0, stream>>>(emb, Wih0, bih0, P0f);
  fill_zero_bf16_kernel<<<512, 256, 0, stream>>>(h1a, BATCH * HID);
  fill_zero_bf16_kernel<<<512, 256, 0, stream>>>(h2a, BATCH * HID);

  // ---- recurrence (stream order provides the step dependency) ----
  bf16_t* h1_cur = h1a; bf16_t* h1_nxt = h1b;
  bf16_t* h2_cur = h2a; bf16_t* h2_nxt = h2b;
  const dim3 grid(4, 8), block(256);
  for (int t = 0; t < SEQ; ++t) {
    // layer 0: h1_nxt = tanh(P0[x[:,t]] + h1_cur @ Whh0^T)
    rnn_step_kernel<<<grid, block, 0, stream>>>(
        h1_cur, Whh0b, nullptr, nullptr, P0f, x + t, nullptr, h1_nxt, nullptr);
    // layer 1: h2_nxt = tanh(h1_nxt @ Wih1^T + b_ih1 + h2_cur @ Whh1^T)
    rnn_step_kernel<<<grid, block, 0, stream>>>(
        h1_nxt, Wih1b, h2_cur, Whh1b, nullptr, nullptr, bih1, h2_nxt,
        outs + (size_t)t * BATCH * HID);
    bf16_t* tmp;
    tmp = h1_cur; h1_cur = h1_nxt; h1_nxt = tmp;
    tmp = h2_cur; h2_cur = h2_nxt; h2_nxt = tmp;
  }

  // ---- logits ----
  logits_kernel<<<dim3(1, (SEQ * BATCH) / 64), block, 0, stream>>>(
      outs, fcwb, fcb, out);

  // ---- final states (f32) appended after logits ----
  const size_t logit_elems = (size_t)BATCH * SEQ * VOCAB;
  cvt_bf16_f32_kernel<<<512, 256, 0, stream>>>(
      h1_cur, out + logit_elems, BATCH * HID);
  cvt_bf16_f32_kernel<<<512, 256, 0, stream>>>(
      h2_cur, out + logit_elems + (size_t)BATCH * HID, BATCH * HID);
}
